// ROIPoolingLayer_34694745817119
// MI455X (gfx1250) — matrix operand order repaired and checked by reading, not compiled, not run on hardware
//
#include <hip/hip_runtime.h>
#include <stdint.h>

// CDNA5 / gfx1250 wave32 WMMA types
typedef __attribute__((ext_vector_type(16))) _Float16 v16h;
typedef __attribute__((ext_vector_type(8)))  float    v8f;

#define B_   2
#define N_   2000
#define G_   100
#define H_   1024
#define W_   1024
#define T_   200
#define PCAP 66     // int(200*0.33)
#define NCAP 134    // T - PCAP
#define MH   28
#define MW   28

__device__ __forceinline__ float iou_box(const float p[4], const float g[4]) {
  float yi1 = fmaxf(p[0], g[0]);
  float xi1 = fmaxf(p[1], g[1]);
  float yi2 = fminf(p[2], g[2]);
  float xi2 = fminf(p[3], g[3]);
  float inter = fmaxf(yi2 - yi1, 0.f) * fmaxf(xi2 - xi1, 0.f);
  float a1 = (p[2] - p[0]) * (p[3] - p[1]);
  float a2 = (g[2] - g[0]) * (g[3] - g[1]);
  float u = a1 + a2 - inter;
  return inter / fmaxf(u, 1e-12f);
}

// A-operand fragment (16x32 f16, MxK) from row-major LDS array, per ISA
// table: lanes 0-15 hold K=0..7 (v0..3) and K=16..23 (v4..7); lanes 16-31
// hold K=8..15 and K=24..31, for row M = lane%16.
__device__ __forceinline__ v16h load_fragA(const _Float16* m, int row0, int k0, int stride) {
  int lane = threadIdx.x & 31;
  int l = lane & 15, hi = lane >> 4;
  const _Float16* p = m + (row0 + l) * stride + k0 + hi * 8;
  v16h a;
#pragma unroll
  for (int e = 0; e < 8; ++e) { a[e] = p[e]; a[8 + e] = p[16 + e]; }
  return a;
}

// B-operand fragment (32x16 f16, KxN). Source stored transposed: m[n][k].
// Same K layout as A: lanes 0-15 hold K=0..7 and K=16..23; lanes 16-31
// hold K=8..15 and K=24..31, for column N = lane%16.
__device__ __forceinline__ v16h load_fragB(const _Float16* m, int n0, int k0, int stride) {
  int lane = threadIdx.x & 31;
  int l = lane & 15, hi = lane >> 4;
  const _Float16* p = m + (n0 + l) * stride + k0 + hi * 8;
  v16h b;
#pragma unroll
  for (int e = 0; e < 8; ++e) { b[e] = p[e]; b[8 + e] = p[16 + e]; }
  return b;
}

__global__ void zero_kernel(float* __restrict__ out, int n) {
  int i = blockIdx.x * blockDim.x + threadIdx.x;
  if (i < n) out[i] = 0.f;
}

__global__ __launch_bounds__(256)
void detection_target_kernel(const float* __restrict__ props,      // [B,N,4]
                             const int*   __restrict__ gtcls,      // [B,G]
                             const float* __restrict__ gtbox,      // [B,G,4]
                             const uint8_t* __restrict__ gtmask,   // [B,H,W,G] bool
                             float* __restrict__ out) {
  const int b    = blockIdx.x;
  const int tid  = threadIdx.x;
  const int lane = tid & 31;
  const int wave = tid >> 5;            // 8 waves (wave32)

  __shared__ float    s_gt[G_][4];
  __shared__ int      s_cls[G_];
  __shared__ uint8_t  s_nc[G_], s_cr[G_];
  __shared__ uint8_t  s_pos[N_], s_neg[N_];
  __shared__ int      s_pidx[PCAP], s_nidx[NCAP];
  __shared__ int      s_assign[PCAP];
  __shared__ int      s_numpos, s_numneg;
  __shared__ int      s_row[64], s_col[64];
  __shared__ _Float16 sA[32 * 64];      // A' row weights (28x56 padded)
  __shared__ _Float16 sB[32 * 64];      // B' col weights, stored [n][k]
  __shared__ _Float16 sM[64 * 64];      // gathered mask patch (56x56 padded)
  __shared__ _Float16 sP[32 * 64];      // P = M*B^T, stored transposed [n][k]

  const float*   Pb = props + (size_t)b * N_ * 4;
  const float*   Gb = gtbox + (size_t)b * G_ * 4;
  const int*     Cb = gtcls + b * G_;
  const uint8_t* Mb = gtmask + (size_t)b * H_ * W_ * G_;

  float* out_rois  = out;                                   // [B,T,4]
  int*   out_cls   = (int*)out + B_ * T_ * 4;               // [B,T] (int32 bits)
  float* out_delta = out + B_ * T_ * 4 + B_ * T_;           // [B,T,4]
  float* out_mask  = out + B_ * T_ * 4 + B_ * T_ + B_ * T_ * 4; // [B,T,28,28]

  // ---- GT preprocessing ----
  for (int j = tid; j < G_; j += blockDim.x) {
    float g0 = Gb[j * 4 + 0], g1 = Gb[j * 4 + 1], g2 = Gb[j * 4 + 2], g3 = Gb[j * 4 + 3];
    s_gt[j][0] = g0; s_gt[j][1] = g1; s_gt[j][2] = g2; s_gt[j][3] = g3;
    int c = Cb[j];
    s_cls[j] = c;
    bool v = (fabsf(g0) + fabsf(g1) + fabsf(g2) + fabsf(g3)) > 0.f;
    s_nc[j] = (v && c > 0) ? 1 : 0;
    s_cr[j] = (v && c < 0) ? 1 : 0;
  }
  __syncthreads();

  // ---- per-proposal pos/neg flags ----
  for (int i = tid; i < N_; i += blockDim.x) {
    float p[4] = {Pb[i * 4 + 0], Pb[i * 4 + 1], Pb[i * 4 + 2], Pb[i * 4 + 3]};
    bool v = (fabsf(p[0]) + fabsf(p[1]) + fabsf(p[2]) + fabsf(p[3])) > 0.f;
    float rmax = -1.f, cmax = -1.f;
    for (int j = 0; j < G_; ++j) {
      float o = iou_box(p, s_gt[j]);
      if (s_nc[j]) rmax = fmaxf(rmax, o);
      if (s_cr[j]) cmax = fmaxf(cmax, o);
    }
    s_pos[i] = (v && rmax >= 0.5f) ? 1 : 0;
    s_neg[i] = (v && rmax < 0.5f && cmax < 0.001f) ? 1 : 0;
  }
  __syncthreads();

  // ---- stable selection (order-dependent => serial scan; 2000 elems, trivial) ----
  if (tid == 0) {
    int tp = 0, tn = 0;
    for (int i = 0; i < N_; ++i) {
      if (s_pos[i]) { if (tp < PCAP) s_pidx[tp] = i; ++tp; }
      if (s_neg[i]) { if (tn < NCAP) s_nidx[tn] = i; ++tn; }
    }
    int np = tp < PCAP ? tp : PCAP;
    const float RINV = (float)(1.0 / 0.33);
    int ntarget = (int)(RINV * (float)np) - np;
    int nn = ntarget;
    if (tn < nn) nn = tn;
    if (nn > NCAP) nn = NCAP;
    s_numpos = np;
    s_numneg = nn;
    for (int k = (tp < PCAP ? tp : PCAP); k < PCAP; ++k) s_pidx[k] = 0;
    for (int k = (tn < NCAP ? tn : NCAP); k < NCAP; ++k) s_nidx[k] = 0;
  }
  __syncthreads();
  const int numpos = s_numpos;
  const int numneg = s_numneg;

  // ---- positive slots: GT assignment (first-max argmax), class, rois, deltas ----
  for (int p = tid; p < PCAP; p += blockDim.x) {
    bool valid = p < numpos;
    int pi = s_pidx[p];
    float pr[4] = {Pb[pi * 4 + 0], Pb[pi * 4 + 1], Pb[pi * 4 + 2], Pb[pi * 4 + 3]};
    float best = -2.f; int bj = 0;
    for (int j = 0; j < G_; ++j) {
      float o = s_nc[j] ? iou_box(pr, s_gt[j]) : -1.f;
      if (o > best) { best = o; bj = j; }
    }
    s_assign[p] = bj;
    out_cls[b * T_ + p] = valid ? s_cls[bj] : 0;
    if (valid) {
      out_rois[(b * T_ + p) * 4 + 0] = pr[0];
      out_rois[(b * T_ + p) * 4 + 1] = pr[1];
      out_rois[(b * T_ + p) * 4 + 2] = pr[2];
      out_rois[(b * T_ + p) * 4 + 3] = pr[3];
      float h = pr[2] - pr[0], w = pr[3] - pr[1];
      float hs = h > 0.f ? h : 1.f, ws = w > 0.f ? w : 1.f;
      float gh = s_gt[bj][2] - s_gt[bj][0], gw = s_gt[bj][3] - s_gt[bj][1];
      float ghs = gh > 0.f ? gh : 1.f, gws = gw > 0.f ? gw : 1.f;
      float cy = pr[0] + 0.5f * h, cx = pr[1] + 0.5f * w;
      float gcy = s_gt[bj][0] + 0.5f * gh, gcx = s_gt[bj][1] + 0.5f * gw;
      out_delta[(b * T_ + p) * 4 + 0] = ((gcy - cy) / hs) / 0.1f;
      out_delta[(b * T_ + p) * 4 + 1] = ((gcx - cx) / ws) / 0.1f;
      out_delta[(b * T_ + p) * 4 + 2] = logf(ghs / hs) / 0.2f;
      out_delta[(b * T_ + p) * 4 + 3] = logf(gws / ws) / 0.2f;
    }
  }
  // ---- negative rois at positions numpos..numpos+numneg ----
  for (int j = tid; j < NCAP; j += blockDim.x) {
    if (j < numneg) {
      int ni = s_nidx[j];
      int t = numpos + j;
      out_rois[(b * T_ + t) * 4 + 0] = Pb[ni * 4 + 0];
      out_rois[(b * T_ + t) * 4 + 1] = Pb[ni * 4 + 1];
      out_rois[(b * T_ + t) * 4 + 2] = Pb[ni * 4 + 2];
      out_rois[(b * T_ + t) * 4 + 3] = Pb[ni * 4 + 3];
    }
  }

  // ---- mask crop+resize as separable bilinear: out = A * M * B^T on WMMA ----
  for (int i = tid; i < 32 * 64; i += blockDim.x) {
    sA[i] = (_Float16)0.f; sB[i] = (_Float16)0.f; sP[i] = (_Float16)0.f;
  }
  for (int i = tid; i < 64 * 64; i += blockDim.x) sM[i] = (_Float16)0.f;
  __syncthreads();

  for (int p = 0; p < PCAP; ++p) {
    if (p >= numpos) break;                 // uniform: numpos shared by block
    int pi = s_pidx[p];
    int g  = s_assign[p];
    float y1 = Pb[pi * 4 + 0], x1 = Pb[pi * 4 + 1];
    float y2 = Pb[pi * 4 + 2], x2 = Pb[pi * 4 + 3];

    if (tid < MH) {                         // interpolation weights (rows)
      int i = tid;
      float yv = y1 * (H_ - 1) + (float)i * ((y2 - y1) * (H_ - 1) / (float)(MH - 1));
      float ty = floorf(yv);
      float yl = yv - ty;
      int tyi = (int)fminf(fmaxf(ty, 0.f), (float)(H_ - 1));
      int byi = (int)fminf(fmaxf(ty + 1.f, 0.f), (float)(H_ - 1));
      s_row[2 * i] = tyi; s_row[2 * i + 1] = byi;
      bool ok = (yv >= 0.f) && (yv <= (float)(H_ - 1));
      sA[i * 64 + 2 * i]     = ok ? (_Float16)(1.f - yl) : (_Float16)0.f;
      sA[i * 64 + 2 * i + 1] = ok ? (_Float16)yl         : (_Float16)0.f;
    } else if (tid >= 32 && tid < 32 + MW) { // interpolation weights (cols)
      int j = tid - 32;
      float xv = x1 * (W_ - 1) + (float)j * ((x2 - x1) * (W_ - 1) / (float)(MW - 1));
      float tx = floorf(xv);
      float xl = xv - tx;
      int txi = (int)fminf(fmaxf(tx, 0.f), (float)(W_ - 1));
      int bxi = (int)fminf(fmaxf(tx + 1.f, 0.f), (float)(W_ - 1));
      s_col[2 * j] = txi; s_col[2 * j + 1] = bxi;
      bool ok = (xv >= 0.f) && (xv <= (float)(W_ - 1));
      sB[j * 64 + 2 * j]     = ok ? (_Float16)(1.f - xl) : (_Float16)0.f;
      sB[j * 64 + 2 * j + 1] = ok ? (_Float16)xl         : (_Float16)0.f;
    }
    __syncthreads();

    // gather 56x56 patch of the assigned bool mask (same bytes the scalar
    // gather method would touch; no extra HBM traffic)
    for (int idx = tid; idx < 56 * 56; idx += blockDim.x) {
      int rr = idx / 56, cc = idx % 56;
      int y = s_row[rr], x = s_col[cc];
      sM[rr * 64 + cc] = (_Float16)(float)(Mb[((size_t)y * W_ + x) * G_ + g] ? 1 : 0);
    }
    __syncthreads();

    // stage 1: P(64x32) = M(64x64) x B'^T ; one 16x16 tile per wave (8 waves)
    {
      int mt = wave >> 1, nt = wave & 1;
      v8f acc = {0.f, 0.f, 0.f, 0.f, 0.f, 0.f, 0.f, 0.f};
#pragma unroll
      for (int kt = 0; kt < 2; ++kt) {
        v16h a  = load_fragA(sM, 16 * mt, 32 * kt, 64);
        v16h bb = load_fragB(sB, 16 * nt, 32 * kt, 64);
        acc = __builtin_amdgcn_wmma_f32_16x16x32_f16(false, a, false, bb,
                                                     (short)0, acc, false, false);
      }
      int l = lane & 15, hi = lane >> 4;
      int col = 16 * nt + l;
#pragma unroll
      for (int r = 0; r < 8; ++r) {
        int row = 16 * mt + hi * 8 + r;
        sP[col * 64 + row] = (_Float16)acc[r];   // store transposed: [n][k]
      }
    }
    __syncthreads();

    // stage 2: OUT(32x32) = A'(32x64) x P(64x32) ; waves 0..3
    if (wave < 4) {
      int mt = wave >> 1, nt = wave & 1;
      v8f acc = {0.f, 0.f, 0.f, 0.f, 0.f, 0.f, 0.f, 0.f};
#pragma unroll
      for (int kt = 0; kt < 2; ++kt) {
        v16h a  = load_fragA(sA, 16 * mt, 32 * kt, 64);
        v16h bb = load_fragB(sP, 16 * nt, 32 * kt, 64);
        acc = __builtin_amdgcn_wmma_f32_16x16x32_f16(false, a, false, bb,
                                                     (short)0, acc, false, false);
      }
      int l = lane & 15, hi = lane >> 4;
      int col = 16 * nt + l;
      if (col < MW) {
#pragma unroll
        for (int r = 0; r < 8; ++r) {
          int row = 16 * mt + hi * 8 + r;
          if (row < MH)
            out_mask[(((size_t)b * T_ + p) * MH + row) * MW + col] = rintf(acc[r]);
        }
      }
    }
    __syncthreads();
  }
}

extern "C" void kernel_launch(void* const* d_in, const int* in_sizes, int n_in,
                              void* d_out, int out_size, void* d_ws, size_t ws_size,
                              hipStream_t stream) {
  const float*   props  = (const float*)d_in[0];   // proposals [2,2000,4] f32
  const int*     gtcls  = (const int*)d_in[1];     // gt_class_ids [2,100] i32
  const float*   gtbox  = (const float*)d_in[2];   // gt_boxes [2,100,4] f32
  const uint8_t* gtmask = (const uint8_t*)d_in[3]; // gt_masks [2,1024,1024,100] bool
  float* out = (float*)d_out;

  // Harness poisons d_out; reference output is mostly zeros -> zero-init.
  zero_kernel<<<(out_size + 255) / 256, 256, 0, stream>>>(out, out_size);
  // One workgroup (8 wave32 waves) per batch; everything staged in LDS.
  detection_target_kernel<<<dim3(B_), dim3(256), 0, stream>>>(props, gtcls, gtbox, gtmask, out);
}
